// SparseMoE_52561809768845
// MI455X (gfx1250) — compile-verified, matrix operand order checked by reading, not run on hardware
//
#include <hip/hip_runtime.h>
#include <cstdint>
#include <cstddef>

// Problem constants (from reference): B=2, S=2048, D=1024, E=8, H=4096, top_k=2
#define B_    2
#define S_    2048
#define D_    1024
#define E_    8
#define H_    4096
#define NTOK  (B_ * S_)          // 4096 tokens
#define NSLOT (NTOK * 2)         // 8192 routed slots (top_k = 2, indices distinct)
#define NTILE_MAX 136            // sum_e ceil(c_e/64) <= 8192/64 + 8
#define PADCAP (NTILE_MAX * 64)  // 8704 padded slot rows
#define TN    128                // GEMM column tile
#define TKS   32                 // GEMM K step (bf16 WMMA depth)

typedef __bf16 bf16;
typedef bf16  bf16x2  __attribute__((ext_vector_type(2)));
typedef bf16  bf16x4  __attribute__((ext_vector_type(4)));
typedef bf16  bf16x8  __attribute__((ext_vector_type(8)));
typedef bf16  bf16x16 __attribute__((ext_vector_type(16)));
typedef float f32x8   __attribute__((ext_vector_type(8)));

// ---------------------------------------------------------------------------
// Fragment load following CDNA5 ISA 16-bit A/B layout (§7.12.2):
// lane L holds row/col (L&15); elements 0..7 = K in [k0, k0+8), elements
// 8..15 = K in [k0+16, k0+24), with k0 = 0 (lanes 0-15) or 8 (lanes 16-31).
// Row stride is 40 halfs (80B, 16B aligned) so each chunk is one ds_load_b128.
// ---------------------------------------------------------------------------
__device__ __forceinline__ bf16x16 load_frag(const bf16* rowbase, int k0) {
  const bf16x8* p = (const bf16x8*)rowbase;
  bf16x8 lo = p[k0 >> 3];
  bf16x8 hi = p[(k0 >> 3) + 2];
  bf16x16 r;
#pragma unroll
  for (int i = 0; i < 8; ++i) { r[i] = lo[i]; r[8 + i] = hi[i]; }
  return r;
}

// Branch-free tanh-GELU: 0.5*v*(1+tanh(z)) == v * sigmoid(2z),
// 2z = 1.5957691216*(v + 0.044715 v^3). One v_exp_f32 + v_rcp_f32.
__device__ __forceinline__ float gelu_tanh(float v) {
  float u = 1.5957691216057308f * v * __builtin_fmaf(0.044715f, v * v, 1.0f);
  return v * __builtin_amdgcn_rcpf(1.0f + __expf(-u));
}

// ---------------------------------------------------------------------------
// K0: init — zero out-region, counters; slot->token map = -1
// ---------------------------------------------------------------------------
__global__ __launch_bounds__(256) void init_kernel(float* __restrict__ out,
                                                   int* __restrict__ counts,
                                                   int* __restrict__ cursor,
                                                   int* __restrict__ slot_tok) {
  int i = blockIdx.x * 256 + threadIdx.x;
  if (i < NTOK * D_) out[i] = 0.0f;
  if (i < E_) { counts[i] = 0; cursor[i] = 0; }
  if (i < PADCAP) slot_tok[i] = -1;
}

// ---------------------------------------------------------------------------
// K1: x (f32) -> bf16, vectorized
// ---------------------------------------------------------------------------
__global__ __launch_bounds__(256) void cvt_kernel(const float* __restrict__ x,
                                                  bf16* __restrict__ xb) {
  int i = blockIdx.x * 256 + threadIdx.x;  // over (NTOK*D_)/4
  float4 v = ((const float4*)x)[i];
  bf16x4 o;
  o[0] = (bf16)v.x; o[1] = (bf16)v.y; o[2] = (bf16)v.z; o[3] = (bf16)v.w;
  ((bf16x4*)xb)[i] = o;
}

// ---------------------------------------------------------------------------
// K2: router — one wave32 per token. Lanes split D; 16 f32 accumulators per
// lane (8 route + 8 noise), shfl_xor reduction, then lane 0 does softplus /
// noisy / softmax (gate1) / top-2.
// ---------------------------------------------------------------------------
__global__ __launch_bounds__(256) void router_kernel(
    const float* __restrict__ x, const float* __restrict__ noise,
    const float* __restrict__ w_route, const float* __restrict__ b_route,
    const float* __restrict__ w_noise, const float* __restrict__ b_noise,
    float* __restrict__ gate1, int* __restrict__ topk_e,
    float* __restrict__ topk_w, int* __restrict__ counts) {
  int gtid = blockIdx.x * 256 + threadIdx.x;
  int tok  = gtid >> 5;
  int lane = threadIdx.x & 31;
  if (tok >= NTOK) return;
  const float* xr = x + (size_t)tok * D_;
  float aR[E_] = {}, aN[E_] = {};
  for (int d = lane; d < D_; d += 32) {
    float xv = xr[d];
    const float4* wr = (const float4*)(w_route + (size_t)d * E_);
    const float4* wn = (const float4*)(w_noise + (size_t)d * E_);
    float4 r0 = wr[0], r1 = wr[1], n0 = wn[0], n1 = wn[1];
    aR[0] += xv * r0.x; aR[1] += xv * r0.y; aR[2] += xv * r0.z; aR[3] += xv * r0.w;
    aR[4] += xv * r1.x; aR[5] += xv * r1.y; aR[6] += xv * r1.z; aR[7] += xv * r1.w;
    aN[0] += xv * n0.x; aN[1] += xv * n0.y; aN[2] += xv * n0.z; aN[3] += xv * n0.w;
    aN[4] += xv * n1.x; aN[5] += xv * n1.y; aN[6] += xv * n1.z; aN[7] += xv * n1.w;
  }
#pragma unroll
  for (int e = 0; e < E_; ++e) {
#pragma unroll
    for (int s = 16; s > 0; s >>= 1) {
      aR[e] += __shfl_xor(aR[e], s, 32);
      aN[e] += __shfl_xor(aN[e], s, 32);
    }
  }
  if (lane == 0) {
    float noisy[E_], mx = -3.0e38f;
#pragma unroll
    for (int e = 0; e < E_; ++e) {
      float lg = aR[e] + b_route[e];
      float nl = aN[e] + b_noise[e];
      float sp = (nl > 20.0f) ? nl : log1pf(__expf(nl));  // softplus
      noisy[e] = lg + noise[(size_t)tok * E_ + e] * sp;
      mx = fmaxf(mx, noisy[e]);
    }
    float ex[E_], s = 0.0f;
#pragma unroll
    for (int e = 0; e < E_; ++e) { ex[e] = __expf(noisy[e] - mx); s += ex[e]; }
    float inv = 1.0f / s;
#pragma unroll
    for (int e = 0; e < E_; ++e) gate1[(size_t)tok * E_ + e] = ex[e] * inv;
    // top-2 (first-occurrence on ties, matching lax.top_k)
    int i1 = 0;
#pragma unroll
    for (int e = 1; e < E_; ++e) if (noisy[e] > noisy[i1]) i1 = e;
    int i2 = (i1 == 0) ? 1 : 0;
#pragma unroll
    for (int e = 0; e < E_; ++e) if (e != i1 && noisy[e] > noisy[i2]) i2 = e;
    float e2 = __expf(noisy[i2] - noisy[i1]);
    float sw = 1.0f + e2;
    topk_e[tok * 2 + 0] = i1; topk_e[tok * 2 + 1] = i2;
    topk_w[tok * 2 + 0] = 1.0f / sw; topk_w[tok * 2 + 1] = e2 / sw;
    atomicAdd(&counts[i1], 1);
    atomicAdd(&counts[i2], 1);
  }
}

// ---------------------------------------------------------------------------
// K3: 64-aligned per-expert segment offsets + tile -> expert map
// ---------------------------------------------------------------------------
__global__ void offsets_kernel(const int* __restrict__ counts,
                               int* __restrict__ poff,
                               int* __restrict__ tile_expert) {
  if (threadIdx.x != 0 || blockIdx.x != 0) return;
  int t = 0, off = 0;
  for (int e = 0; e < E_; ++e) {
    poff[e] = off;
    int tiles = (counts[e] + 63) >> 6;
    for (int i = 0; i < tiles; ++i) tile_expert[t++] = e;
    off += tiles * 64;
  }
  poff[E_] = off;
  for (; t < NTILE_MAX; ++t) tile_expert[t] = -1;
}

// ---------------------------------------------------------------------------
// K4: scatter (token,k) into expert-contiguous slots
// ---------------------------------------------------------------------------
__global__ __launch_bounds__(256) void scatter_kernel(
    const int* __restrict__ topk_e, const float* __restrict__ topk_w,
    const int* __restrict__ poff, int* __restrict__ cursor,
    int* __restrict__ slot_tok, float* __restrict__ slot_w) {
  int i = blockIdx.x * 256 + threadIdx.x;
  if (i >= NSLOT) return;
  int e = topk_e[i];
  int pos = atomicAdd(&cursor[e], 1);
  int slot = poff[e] + pos;
  slot_tok[slot] = i >> 1;
  slot_w[slot]   = topk_w[i];
}

// ---------------------------------------------------------------------------
// Shared GEMM tile machinery: 64 rows x TN(=128) cols, K step 32.
// 128 threads = 4 waves; wave w owns rows [16w,16w+16) x TN cols
// = 8 accumulators of v_wmma_f32_16x16x32_bf16 per K step.
// B staging: thread t handles cols {2c, 2c+1} (c = t&63) over 16 K values
// (t>>6 selects K half): float2 global loads (coalesced), v_cvt_pk_bf16_f32
// pairs, ds_store_b32.
// ---------------------------------------------------------------------------
__device__ __forceinline__ void stage_B(const float* __restrict__ wtile,
                                        int ldw, int tid,
                                        bf16 (*Bs)[TKS + 8]) {
  int nn0 = (tid & 63) * 2;
  int kdb = (tid >> 6) * 16;  // 0 or 16
  const float* wk = wtile + (size_t)kdb * ldw + nn0;
#pragma unroll
  for (int j = 0; j < 8; ++j) {
    float2 r0 = *(const float2*)(wk + (size_t)(2 * j) * ldw);
    float2 r1 = *(const float2*)(wk + (size_t)(2 * j + 1) * ldw);
    bf16x2 p0; p0[0] = (bf16)r0.x; p0[1] = (bf16)r1.x;
    bf16x2 p1; p1[0] = (bf16)r0.y; p1[1] = (bf16)r1.y;
    *(bf16x2*)(&Bs[nn0 + 0][kdb + 2 * j]) = p0;
    *(bf16x2*)(&Bs[nn0 + 1][kdb + 2 * j]) = p1;
  }
}

// ---------------------------------------------------------------------------
// K5: GEMM1  h[slot, :] = gelu(x[tok] @ w_fc[e] + b_fc[e])   (bf16 WMMA)
// ---------------------------------------------------------------------------
__global__ __launch_bounds__(128) void gemm_fc_kernel(
    const bf16* __restrict__ xb, const float* __restrict__ w_fc,
    const float* __restrict__ b_fc, const int* __restrict__ slot_tok,
    const int* __restrict__ tile_expert, bf16* __restrict__ h) {
  int e = tile_expert[blockIdx.y];
  if (e < 0) return;
  int r0 = blockIdx.y * 64;
  int h0 = blockIdx.x * TN;
  __shared__ bf16 As[64][TKS + 8];
  __shared__ bf16 Bs[TN][TKS + 8];
  int tid  = threadIdx.x;
  int wv   = tid >> 5;
  int lane = tid & 31;
  int arow = tid >> 1;
  int aoff = (tid & 1) * 16;
  int tokA = slot_tok[r0 + arow];
  const float* wbase = w_fc + (size_t)e * D_ * H_ + h0;

  f32x8 acc[8] = {};
  for (int kb = 0; kb < D_; kb += TKS) {
    // A: 64x32 bf16 (gathered token rows; padded rows load zero)
    {
      bf16x8 v0 = {}, v1 = {};
      if (tokA >= 0) {
        const bf16x8* src = (const bf16x8*)(xb + (size_t)tokA * D_ + kb + aoff);
        v0 = src[0]; v1 = src[1];
      }
      bf16x8* dst = (bf16x8*)(&As[arow][aoff]);
      dst[0] = v0; dst[1] = v1;
    }
    stage_B(wbase + (size_t)kb * H_, H_, tid, Bs);
    __syncthreads();
    int k0 = (lane < 16) ? 0 : 8;
    bf16x16 a = load_frag(&As[wv * 16 + (lane & 15)][0], k0);
#pragma unroll
    for (int j = 0; j < 8; ++j) {
      bf16x16 b = load_frag(&Bs[j * 16 + (lane & 15)][0], k0);
      acc[j] = __builtin_amdgcn_wmma_f32_16x16x32_bf16(
          false, a, false, b, (short)0, acc[j], false, false);
    }
    __syncthreads();
  }
  // Epilogue: +bias, tanh-GELU, store bf16 h.
  int rbase = r0 + wv * 16 + ((lane & 16) ? 8 : 0);
  int col16 = lane & 15;
#pragma unroll
  for (int j = 0; j < 8; ++j) {
    int col = h0 + j * 16 + col16;
    float bias = b_fc[e * H_ + col];
#pragma unroll
    for (int r = 0; r < 8; ++r) {
      float v = acc[j][r] + bias;
      h[(size_t)(rbase + r) * H_ + col] = (bf16)gelu_tanh(v);
    }
  }
}

// ---------------------------------------------------------------------------
// K6: GEMM2  y = h @ w_proj[e] + b_proj[e]; out[tok] += gate * y (atomic f32,
// exactly two commutative adds per element -> deterministic).
// ---------------------------------------------------------------------------
__global__ __launch_bounds__(128) void gemm_proj_kernel(
    const bf16* __restrict__ h, const float* __restrict__ w_proj,
    const float* __restrict__ b_proj, const int* __restrict__ slot_tok,
    const float* __restrict__ slot_w, const int* __restrict__ tile_expert,
    float* __restrict__ out) {
  int e = tile_expert[blockIdx.y];
  if (e < 0) return;
  int r0 = blockIdx.y * 64;
  int d0 = blockIdx.x * TN;
  __shared__ bf16 As[64][TKS + 8];
  __shared__ bf16 Bs[TN][TKS + 8];
  int tid  = threadIdx.x;
  int wv   = tid >> 5;
  int lane = tid & 31;
  int arow = tid >> 1;
  int aoff = (tid & 1) * 16;
  const bf16* hrow = h + (size_t)(r0 + arow) * H_ + aoff;
  const float* wbase = w_proj + (size_t)e * H_ * D_ + d0;

  f32x8 acc[8] = {};
  for (int kb = 0; kb < H_; kb += TKS) {
    {
      const bf16x8* src = (const bf16x8*)(hrow + kb);
      bf16x8* dst = (bf16x8*)(&As[arow][aoff]);
      dst[0] = src[0]; dst[1] = src[1];
    }
    stage_B(wbase + (size_t)kb * D_, D_, tid, Bs);
    __syncthreads();
    int k0 = (lane < 16) ? 0 : 8;
    bf16x16 a = load_frag(&As[wv * 16 + (lane & 15)][0], k0);
#pragma unroll
    for (int j = 0; j < 8; ++j) {
      bf16x16 b = load_frag(&Bs[j * 16 + (lane & 15)][0], k0);
      acc[j] = __builtin_amdgcn_wmma_f32_16x16x32_bf16(
          false, a, false, b, (short)0, acc[j], false, false);
    }
    __syncthreads();
  }
  int rbase = r0 + wv * 16 + ((lane & 16) ? 8 : 0);
  int col16 = lane & 15;
  int   toks[8];
  float sws[8];
#pragma unroll
  for (int r = 0; r < 8; ++r) { toks[r] = slot_tok[rbase + r]; sws[r] = slot_w[rbase + r]; }
#pragma unroll
  for (int j = 0; j < 8; ++j) {
    int col = d0 + j * 16 + col16;
    float bias = b_proj[e * D_ + col];
#pragma unroll
    for (int r = 0; r < 8; ++r) {
      if (toks[r] >= 0) {
        atomicAdd(&out[(size_t)toks[r] * D_ + col], sws[r] * (acc[j][r] + bias));
      }
    }
  }
}

// ---------------------------------------------------------------------------
// Host launch
// ---------------------------------------------------------------------------
extern "C" void kernel_launch(void* const* d_in, const int* in_sizes, int n_in,
                              void* d_out, int out_size, void* d_ws, size_t ws_size,
                              hipStream_t stream) {
  const float* x       = (const float*)d_in[0];
  const float* noise   = (const float*)d_in[1];
  const float* w_route = (const float*)d_in[2];
  const float* b_route = (const float*)d_in[3];
  const float* w_noise = (const float*)d_in[4];
  const float* b_noise = (const float*)d_in[5];
  const float* w_fc    = (const float*)d_in[6];
  const float* b_fc    = (const float*)d_in[7];
  const float* w_proj  = (const float*)d_in[8];
  const float* b_proj  = (const float*)d_in[9];
  // d_in[10] = top_k (fixed at 2 per reference)
  (void)in_sizes; (void)n_in; (void)out_size; (void)ws_size;

  float* out   = (float*)d_out;                // [B*S*D]
  float* gate1 = out + (size_t)NTOK * D_;      // [B*S*E]

  char* ws = (char*)d_ws;
  size_t o = 0;
  auto alloc = [&](size_t bytes) {
    size_t r = o;
    o += (bytes + 255) & ~(size_t)255;
    return r;
  };
  bf16*  xb   = (bf16*) (ws + alloc((size_t)NTOK * D_ * 2));   // ~8 MB
  bf16*  hbuf = (bf16*) (ws + alloc((size_t)PADCAP * H_ * 2)); // ~71 MB
  int*   tke  = (int*)  (ws + alloc((size_t)NSLOT * 4));
  float* tkw  = (float*)(ws + alloc((size_t)NSLOT * 4));
  int*   cnts = (int*)  (ws + alloc(E_ * 4));
  int*   curs = (int*)  (ws + alloc(E_ * 4));
  int*   poff = (int*)  (ws + alloc((E_ + 1) * 4));
  int*   stok = (int*)  (ws + alloc((size_t)PADCAP * 4));
  float* swt  = (float*)(ws + alloc((size_t)PADCAP * 4));
  int*   texp = (int*)  (ws + alloc(NTILE_MAX * 4));

  init_kernel<<<(NTOK * D_ + 255) / 256, 256, 0, stream>>>(out, cnts, curs, stok);
  cvt_kernel<<<(NTOK * D_ / 4 + 255) / 256, 256, 0, stream>>>(x, xb);
  router_kernel<<<(NTOK * 32 + 255) / 256, 256, 0, stream>>>(
      x, noise, w_route, b_route, w_noise, b_noise, gate1, tke, tkw, cnts);
  offsets_kernel<<<1, 32, 0, stream>>>(cnts, poff, texp);
  scatter_kernel<<<(NSLOT + 255) / 256, 256, 0, stream>>>(tke, tkw, poff, curs, stok, swt);
  gemm_fc_kernel<<<dim3(H_ / TN, NTILE_MAX), 128, 0, stream>>>(
      xb, w_fc, b_fc, stok, texp, hbuf);
  gemm_proj_kernel<<<dim3(D_ / TN, NTILE_MAX), 128, 0, stream>>>(
      hbuf, w_proj, b_proj, stok, swt, texp, out);
}
